// SpatialAttention_5738076307518
// MI455X (gfx1250) — matrix operand order repaired and checked by reading, not compiled, not run on hardware
//
#include <hip/hip_runtime.h>
#include <hip/hip_bf16.h>

typedef __attribute__((ext_vector_type(16))) _Float16 v16h;
typedef __attribute__((ext_vector_type(8)))  _Float16 v8h;
typedef __attribute__((ext_vector_type(8)))  float    v8f;

#define B_     8
#define C_     512
#define N_     1024
#define HEADS_ 8
#define D_     64
#define G_     8
#define CPG_   64
#define C3_    1536

union AFrag { v16h v; v8h h[2]; };

__device__ __forceinline__ v8f wmma16(v16h a, v16h b, v8f c) {
  return __builtin_amdgcn_wmma_f32_16x16x32_f16(false, a, false, b, (short)0, c, false, false);
}

// ---------------------------------------------------------------------------
// Kernel 0: convert weights f32 -> f16 in workspace
// ---------------------------------------------------------------------------
__global__ void wcvt_kernel(const float* __restrict__ w1, const float* __restrict__ w2,
                            _Float16* __restrict__ o1, _Float16* __restrict__ o2) {
  int i = blockIdx.x * 256 + threadIdx.x;
  if (i < C3_ * C_) o1[i] = (_Float16)w1[i];
  if (i < C_ * C_)  o2[i] = (_Float16)w2[i];
}

// ---------------------------------------------------------------------------
// Kernel 1: GroupNorm, write hT[b][n][c] (channel-contiguous) as f16
// ---------------------------------------------------------------------------
__global__ void gnorm_kernel(const float* __restrict__ x,
                             const float* __restrict__ gw, const float* __restrict__ gb,
                             _Float16* __restrict__ hT) {
  __shared__ float rs[256], rq[256];
  const int t = threadIdx.x;
  const int b = blockIdx.x >> 3, g = blockIdx.x & 7;
  const float* xg = x + ((size_t)b * C_ + g * CPG_) * N_;   // 64 ch x 1024
  float s = 0.f, ss = 0.f;
  for (int i = 0; i < 256; ++i) {
    float xv = xg[t + i * 256];
    s += xv; ss += xv * xv;
  }
  rs[t] = s; rq[t] = ss;
  __syncthreads();
  for (int off = 128; off > 0; off >>= 1) {
    if (t < off) { rs[t] += rs[t + off]; rq[t] += rq[t + off]; }
    __syncthreads();
  }
  const float inv = 1.0f / (float)(CPG_ * N_);
  const float mean = rs[0] * inv;
  const float var  = rq[0] * inv - mean * mean;
  const float rstd = rsqrtf(var + 1e-5f);

  const int c = t & 63, nsub = t >> 6;                      // 64 ch x 4 n-phases
  const float* xc = xg + (size_t)c * N_;
  const float wv = gw[g * CPG_ + c], bv = gb[g * CPG_ + c];
  _Float16* hb = hT + (size_t)b * N_ * C_ + g * CPG_ + c;
  for (int i = 0; i < 256; ++i) {
    int n = i * 4 + nsub;
    float hv = (xc[n] - mean) * rstd * wv + bv;
    hb[(size_t)n * C_] = (_Float16)hv;
  }
}

// ---------------------------------------------------------------------------
// Kernel 2: QKV GEMM  out[o,n] = sum_c W[o,c] * hT[n,c]  (per batch)
//   q,k stored transposed [bh][n][d]; v natural [bh][d][n]; all f16
// ---------------------------------------------------------------------------
__global__ void qkv_kernel(const _Float16* __restrict__ W,   // [1536][512]
                           const _Float16* __restrict__ hT,  // [B][N][C]
                           const float* __restrict__ bias,   // [1536]
                           _Float16* __restrict__ qT,        // [B*H][N][D]
                           _Float16* __restrict__ kT,        // [B*H][N][D]
                           _Float16* __restrict__ vN) {      // [B*H][D][N]
  const int lane = threadIdx.x & 31, wave = threadIdx.x >> 5;
  const int lo = lane & 15, hi = lane >> 4;
  const int mt = blockIdx.x * 4 + wave;        // 0..95  -> o0
  const int o0 = mt * 16;
  const int n0 = blockIdx.y * 64;
  const int b  = blockIdx.z;
  const _Float16* Bm = hT + ((size_t)b * N_ + n0) * C_;

  v8f acc[4];
  for (int s = 0; s < 4; ++s) acc[s] = (v8f){0.f,0.f,0.f,0.f,0.f,0.f,0.f,0.f};

  const _Float16* arowbase = W + (size_t)(o0 + lo) * C_ + hi * 16;
  for (int kk = 0; kk < 16; ++kk) {
    const int kbase = kk * 32;
    AFrag a;
    a.h[0] = *(const v8h*)(arowbase + kbase);
    a.h[1] = *(const v8h*)(arowbase + kbase + 8);
    #pragma unroll
    for (int s = 0; s < 4; ++s) {
      const _Float16* bcol = Bm + (size_t)(s * 16 + lo) * C_ + kbase + hi * 16;
      v16h bf = *(const v16h*)bcol;
      acc[s] = wmma16(a.v, bf, acc[s]);
    }
  }

  float bj[8];
  #pragma unroll
  for (int j = 0; j < 8; ++j) bj[j] = bias[o0 + j + 8 * hi];

  #pragma unroll
  for (int s = 0; s < 4; ++s) {
    const int n = n0 + s * 16 + lo;
    #pragma unroll
    for (int j = 0; j < 8; ++j) {
      const int o = o0 + j + 8 * hi;
      const _Float16 val = (_Float16)(acc[s][j] + bj[j]);
      if (o < C_) {                               // Q -> transposed
        int head = o >> 6, d = o & 63;
        qT[((size_t)(b * HEADS_ + head) * N_ + n) * D_ + d] = val;
      } else if (o < 2 * C_) {                    // K -> transposed
        int oo = o - C_; int head = oo >> 6, d = oo & 63;
        kT[((size_t)(b * HEADS_ + head) * N_ + n) * D_ + d] = val;
      } else {                                    // V -> natural [d][n]
        int oo = o - 2 * C_; int head = oo >> 6, d = oo & 63;
        vN[((size_t)(b * HEADS_ + head) * D_ + d) * N_ + n] = val;
      }
    }
  }
}

// ---------------------------------------------------------------------------
// Kernel 3: flash attention, one wave = 16 query rows of one (b,head)
//   S = (Q*0.125) K^T via WMMA; online softmax; O += P V^T via WMMA
//   writes O transposed: oT[b][n][head*64+d] f16
// ---------------------------------------------------------------------------
__global__ void attn_kernel(const _Float16* __restrict__ qT,
                            const _Float16* __restrict__ kT,
                            const _Float16* __restrict__ vN,
                            _Float16* __restrict__ oT) {
  __shared__ _Float16 Plds[4][16 * 32];
  const int lane = threadIdx.x & 31, wave = threadIdx.x >> 5;
  const int lo = lane & 15, hi = lane >> 4;
  const int rt = blockIdx.x * 4 + wave;       // 0..63 row tile
  const int n0 = rt * 16;
  const int bh = blockIdx.y;                  // b*8+head
  const int b = bh >> 3, head = bh & 7;

  const _Float16* Q = qT + (size_t)bh * N_ * D_;
  const _Float16* K = kT + (size_t)bh * N_ * D_;
  const _Float16* V = vN + (size_t)bh * D_ * N_;

  // Q fragments (A-layout), scale folded in (1/sqrt(64) = 0.125, exact in f16)
  AFrag aq[2];
  const _Float16* qrow = Q + (size_t)(n0 + lo) * D_ + hi * 16;
  aq[0].h[0] = *(const v8h*)(qrow +  0);
  aq[0].h[1] = *(const v8h*)(qrow +  8);
  aq[1].h[0] = *(const v8h*)(qrow + 32);
  aq[1].h[1] = *(const v8h*)(qrow + 40);
  aq[0].v = aq[0].v * (_Float16)0.125f;
  aq[1].v = aq[1].v * (_Float16)0.125f;

  float rm[8], rl[8];
  #pragma unroll
  for (int j = 0; j < 8; ++j) { rm[j] = -3.0e38f; rl[j] = 0.f; }
  v8f oacc[4];
  for (int t = 0; t < 4; ++t) oacc[t] = (v8f){0.f,0.f,0.f,0.f,0.f,0.f,0.f,0.f};

  _Float16* pw = &Plds[wave][0];

  for (int mt = 0; mt < 32; ++mt) {
    const int m0 = mt * 32;
    // --- scores: two 16x16 tiles over 32 keys, K-dim = d = 64 (2 slices)
    v8f s0 = (v8f){0.f,0.f,0.f,0.f,0.f,0.f,0.f,0.f};
    v8f s1 = (v8f){0.f,0.f,0.f,0.f,0.f,0.f,0.f,0.f};
    const _Float16* k0 = K + (size_t)(m0 + lo) * D_ + hi * 16;
    const _Float16* k1 = K + (size_t)(m0 + 16 + lo) * D_ + hi * 16;
    s0 = wmma16(aq[0].v, *(const v16h*)(k0),      s0);
    s0 = wmma16(aq[1].v, *(const v16h*)(k0 + 32), s0);
    s1 = wmma16(aq[0].v, *(const v16h*)(k1),      s1);
    s1 = wmma16(aq[1].v, *(const v16h*)(k1 + 32), s1);

    // --- online softmax; row r = j + 8*hi lives in 16 lanes of this half
    float corr[8], p0v[8], p1v[8];
    #pragma unroll
    for (int j = 0; j < 8; ++j) {
      float x0 = s0[j], x1 = s1[j];
      float mx = fmaxf(x0, x1);
      for (int off = 1; off < 16; off <<= 1) mx = fmaxf(mx, __shfl_xor(mx, off, 16));
      float nm = fmaxf(rm[j], mx);
      corr[j] = __expf(rm[j] - nm);
      float p0 = __expf(x0 - nm), p1 = __expf(x1 - nm);
      float ps = p0 + p1;
      for (int off = 1; off < 16; off <<= 1) ps += __shfl_xor(ps, off, 16);
      rl[j] = rl[j] * corr[j] + ps;
      rm[j] = nm;
      p0v[j] = p0; p1v[j] = p1;
    }
    #pragma unroll
    for (int t = 0; t < 4; ++t)
      #pragma unroll
      for (int j = 0; j < 8; ++j) oacc[t][j] = oacc[t][j] * corr[j];

    // --- re-layout P: C-fragment -> LDS [row][col] -> A-fragment
    __syncthreads();
    #pragma unroll
    for (int j = 0; j < 8; ++j) {
      pw[(j + 8 * hi) * 32 + lo]      = (_Float16)p0v[j];
      pw[(j + 8 * hi) * 32 + 16 + lo] = (_Float16)p1v[j];
    }
    __syncthreads();
    AFrag ap;
    const _Float16* prow = pw + lo * 32 + hi * 16;
    ap.h[0] = *(const v8h*)(prow);
    ap.h[1] = *(const v8h*)(prow + 8);

    // --- O += P * V^T  (B columns = v[d][m..] contiguous in m)
    #pragma unroll
    for (int t = 0; t < 4; ++t) {
      const _Float16* vcol = V + (size_t)(t * 16 + lo) * N_ + m0 + hi * 16;
      oacc[t] = wmma16(ap.v, *(const v16h*)vcol, oacc[t]);
    }
  }

  // --- normalize and store transposed for proj GEMM
  #pragma unroll
  for (int t = 0; t < 4; ++t) {
    #pragma unroll
    for (int j = 0; j < 8; ++j) {
      const int n = n0 + j + 8 * hi;
      const int d = t * 16 + lo;
      oT[((size_t)b * N_ + n) * C_ + head * D_ + d] = (_Float16)(oacc[t][j] / rl[j]);
    }
  }
}

// ---------------------------------------------------------------------------
// Kernel 4: proj GEMM + bias + residual -> f32 out
// ---------------------------------------------------------------------------
__global__ void proj_kernel(const _Float16* __restrict__ W,   // [512][512]
                            const _Float16* __restrict__ oT,  // [B][N][C]
                            const float* __restrict__ bias,
                            const float* __restrict__ x,
                            float* __restrict__ out) {
  const int lane = threadIdx.x & 31, wave = threadIdx.x >> 5;
  const int lo = lane & 15, hi = lane >> 4;
  const int o0 = (blockIdx.x * 4 + wave) * 16;   // 0..511
  const int n0 = blockIdx.y * 64;
  const int b  = blockIdx.z;
  const _Float16* Bm = oT + ((size_t)b * N_ + n0) * C_;

  v8f acc[4];
  for (int s = 0; s < 4; ++s) acc[s] = (v8f){0.f,0.f,0.f,0.f,0.f,0.f,0.f,0.f};

  const _Float16* arowbase = W + (size_t)(o0 + lo) * C_ + hi * 16;
  for (int kk = 0; kk < 16; ++kk) {
    const int kbase = kk * 32;
    AFrag a;
    a.h[0] = *(const v8h*)(arowbase + kbase);
    a.h[1] = *(const v8h*)(arowbase + kbase + 8);
    #pragma unroll
    for (int s = 0; s < 4; ++s) {
      const _Float16* bcol = Bm + (size_t)(s * 16 + lo) * C_ + kbase + hi * 16;
      acc[s] = wmma16(a.v, *(const v16h*)bcol, acc[s]);
    }
  }

  float bj[8];
  #pragma unroll
  for (int j = 0; j < 8; ++j) bj[j] = bias[o0 + j + 8 * hi];

  #pragma unroll
  for (int s = 0; s < 4; ++s) {
    const int n = n0 + s * 16 + lo;
    #pragma unroll
    for (int j = 0; j < 8; ++j) {
      const int o = o0 + j + 8 * hi;
      const size_t idx = ((size_t)b * C_ + o) * N_ + n;
      out[idx] = x[idx] + bj[j] + acc[s][j];
    }
  }
}

// ---------------------------------------------------------------------------
// Workspace layout (bytes)
// ---------------------------------------------------------------------------
static constexpr size_t OFF_HT = 0;                                   // 8 MB
static constexpr size_t OFF_W1 = OFF_HT + (size_t)B_ * N_ * C_ * 2;   // 1.5 MB
static constexpr size_t OFF_W2 = OFF_W1 + (size_t)C3_ * C_ * 2;       // 0.5 MB
static constexpr size_t OFF_Q  = OFF_W2 + (size_t)C_ * C_ * 2;        // 8 MB
static constexpr size_t OFF_K  = OFF_Q + (size_t)B_ * C_ * N_ * 2;    // 8 MB
static constexpr size_t OFF_V  = OFF_K + (size_t)B_ * C_ * N_ * 2;    // 8 MB
static constexpr size_t OFF_O  = OFF_V + (size_t)B_ * C_ * N_ * 2;    // 8 MB

extern "C" void kernel_launch(void* const* d_in, const int* in_sizes, int n_in,
                              void* d_out, int out_size, void* d_ws, size_t ws_size,
                              hipStream_t stream) {
  const float* x      = (const float*)d_in[0];
  const float* gn_w   = (const float*)d_in[1];
  const float* gn_b   = (const float*)d_in[2];
  const float* qkv_w  = (const float*)d_in[3];
  const float* qkv_b  = (const float*)d_in[4];
  const float* proj_w = (const float*)d_in[5];
  const float* proj_b = (const float*)d_in[6];
  float* out = (float*)d_out;

  char* ws = (char*)d_ws;
  _Float16* hT  = (_Float16*)(ws + OFF_HT);
  _Float16* w1h = (_Float16*)(ws + OFF_W1);
  _Float16* w2h = (_Float16*)(ws + OFF_W2);
  _Float16* qT  = (_Float16*)(ws + OFF_Q);
  _Float16* kT  = (_Float16*)(ws + OFF_K);
  _Float16* vN  = (_Float16*)(ws + OFF_V);
  _Float16* oT  = (_Float16*)(ws + OFF_O);

  wcvt_kernel<<<(C3_ * C_) / 256, 256, 0, stream>>>(qkv_w, proj_w, w1h, w2h);
  gnorm_kernel<<<B_ * G_, 256, 0, stream>>>(x, gn_w, gn_b, hT);
  qkv_kernel<<<dim3(24, 16, B_), 128, 0, stream>>>(w1h, hT, qkv_b, qT, kT, vN);
  attn_kernel<<<dim3(16, B_ * HEADS_), 128, 0, stream>>>(qT, kT, vN, oT);
  proj_kernel<<<dim3(8, 16, B_), 128, 0, stream>>>(w2h, oT, proj_b, x, out);
}
